// GLM2Block_62929860821273
// MI455X (gfx1250) — compile-verified, matrix operand order checked by reading, not run on hardware
//
#include <hip/hip_runtime.h>
#include <hip/hip_bf16.h>
#include <math.h>

typedef __bf16 bf16;
typedef __attribute__((ext_vector_type(16))) __bf16 v16bf;
typedef __attribute__((ext_vector_type(8)))  __bf16 v8bf;
typedef __attribute__((ext_vector_type(8)))  float  v8f;

#define LOG2E 1.4426950408889634f

union FragU { v16bf v; v8bf h[2]; };

// Load a 16x32 bf16 WMMA A-operand fragment (B^T treated identically) from LDS.
// ISA 16-bit A layout: lanes 0-15 hold row m, K chunks {0..7, 16..23}; lanes 16-31 hold {8..15, 24..31}.
static __device__ inline v16bf ld_frag(const bf16* base, int rowbase, int stride, int k0, int lane) {
    int r  = rowbase + (lane & 15);
    int lo = (lane & 16) ? 8 : 0;
    const bf16* p = base + r * stride + k0 + lo;
    FragU f;
    f.h[0] = *(const v8bf*)(p);
    f.h[1] = *(const v8bf*)(p + 16);
    return f.v;
}

// CDNA5 async global->LDS copy (16B per lane), tracked by ASYNCcnt.
static __device__ inline void async_ld16(unsigned int lds_byte_off, unsigned int g_byte_off,
                                         unsigned long long sbase) {
    asm volatile("global_load_async_to_lds_b128 %0, %1, %2"
                 :: "v"(lds_byte_off), "v"(g_byte_off), "s"(sbase)
                 : "memory");
}
static __device__ inline void wait_async0() {
    asm volatile("s_wait_asynccnt 0" ::: "memory");
}

// fast hardware TRANS ops (co-execute with XDL WMMA)
static __device__ inline float fast_exp2(float x) { return __builtin_amdgcn_exp2f(x); }
static __device__ inline float fast_rcp(float x)  { return __builtin_amdgcn_rcpf(x); }

// ---------------------------------------------------------------------------
// RMSNorm: fp32 in -> bf16 out, one row (4096) per block of 256 threads
// ---------------------------------------------------------------------------
__global__ __launch_bounds__(256)
void rmsnorm_kernel(const float* __restrict__ x, const float* __restrict__ w,
                    bf16* __restrict__ out, int H) {
    const int row = blockIdx.x;
    const int tid = threadIdx.x;
    const float* xr = x + (size_t)row * H;
    float ss = 0.f;
    for (int i = tid; i < H / 4; i += 256) {
        float4 v = ((const float4*)xr)[i];
        ss += v.x * v.x + v.y * v.y + v.z * v.z + v.w * v.w;
    }
    for (int off = 16; off > 0; off >>= 1) ss += __shfl_xor(ss, off, 32);
    __shared__ float red[8];
    if ((tid & 31) == 0) red[tid >> 5] = ss;
    __syncthreads();
    float tot = 0.f;
    for (int i = 0; i < 8; i++) tot += red[i];
    const float r = rsqrtf(tot / (float)H + 1e-5f);
    bf16* orow = out + (size_t)row * H;
    for (int i = tid * 4; i < H; i += 1024) {
        float4 xv = *(const float4*)&xr[i];
        float4 wv = *(const float4*)&w[i];
        orow[i + 0] = (bf16)(wv.x * xv.x * r);
        orow[i + 1] = (bf16)(wv.y * xv.y * r);
        orow[i + 2] = (bf16)(wv.z * xv.z * r);
        orow[i + 3] = (bf16)(wv.w * xv.w * r);
    }
}

// ---------------------------------------------------------------------------
// Generic NT GEMM: C[m][n] = sum_k A_bf16[m][k] * W_f32[n][k] (+bias[n]) (+res[m][n])
// 128x128 tile / WG, 8 waves (32x64 patch each). Double-buffered LDS, one
// barrier per K-step; A staged via async global->LDS, B via reg pipeline.
// ---------------------------------------------------------------------------
template<bool HAS_BIAS, bool HAS_RES>
__global__ __launch_bounds__(256)
void gemm_bf16_kernel(const bf16* __restrict__ A, const float* __restrict__ W,
                      const float* __restrict__ bias, const float* __restrict__ res,
                      float* __restrict__ outF,
                      int M, int N, int K) {
    __shared__ bf16 As[2][128 * 40];
    __shared__ bf16 Bs[2][128 * 40];
    const int tid = threadIdx.x, lane = tid & 31, w = tid >> 5;
    const int wm = w & 3, wn = w >> 2;
    const int mb = blockIdx.x * 128, nb = blockIdx.y * 128;
    const int NIT = K >> 5;

    // this thread's two 16B A chunks within the 128x32 tile
    const int ar0 = tid >> 2,          akc0 = (tid & 3) << 3;
    const int ar1 = (tid + 256) >> 2,  akc1 = ((tid + 256) & 3) << 3;
    unsigned int ldsA0[2], ldsA1[2];
    for (int b = 0; b < 2; b++) {
        ldsA0[b] = (unsigned int)(uintptr_t)&As[b][ar0 * 40 + akc0];
        ldsA1[b] = (unsigned int)(uintptr_t)&As[b][ar1 * 40 + akc1];
    }
    const unsigned long long abase = (unsigned long long)(uintptr_t)A;
    const unsigned int ag0 = (unsigned int)(((size_t)(mb + ar0) * K + akc0) * 2);
    const unsigned int ag1 = (unsigned int)(((size_t)(mb + ar1) * K + akc1) * 2);

    auto issueA = [&](int it, int buf) {
        const unsigned int kb = (unsigned int)it << 6;   // 32 elems * 2B
        async_ld16(ldsA0[buf], ag0 + kb, abase);
        async_ld16(ldsA1[buf], ag1 + kb, abase);
    };

    float4 breg[4];
    auto loadB = [&](int it) {
        const int k0 = it << 5;
        #pragma unroll
        for (int i = 0; i < 4; i++) {
            int ch = tid + 256 * i;
            int r = ch >> 3, kc = (ch & 7) << 2;
            breg[i] = *(const float4*)&W[(size_t)(nb + r) * K + k0 + kc];
        }
    };
    auto storeB = [&](int buf) {
        #pragma unroll
        for (int i = 0; i < 4; i++) {
            int ch = tid + 256 * i;
            int r = ch >> 3, kc = (ch & 7) << 2;
            bf16* d = &Bs[buf][r * 40 + kc];
            d[0] = (bf16)breg[i].x; d[1] = (bf16)breg[i].y;
            d[2] = (bf16)breg[i].z; d[3] = (bf16)breg[i].w;
        }
    };

    v8f c[2][4];
    #pragma unroll
    for (int i = 0; i < 2; i++)
        for (int j = 0; j < 4; j++)
            for (int e = 0; e < 8; e++) c[i][j][e] = 0.f;

    issueA(0, 0);
    loadB(0);
    storeB(0);
    wait_async0();
    __syncthreads();

    for (int it = 0; it < NIT; ++it) {
        const int cur = it & 1;
        if (it + 1 < NIT) {          // prefetch next K-step into other buffer
            issueA(it + 1, cur ^ 1);
            loadB(it + 1);
            storeB(cur ^ 1);
        }
        v16bf af[2], bfr[4];
        #pragma unroll
        for (int i = 0; i < 2; i++) af[i]  = ld_frag(As[cur], 32 * wm + 16 * i, 40, 0, lane);
        #pragma unroll
        for (int j = 0; j < 4; j++) bfr[j] = ld_frag(Bs[cur], 64 * wn + 16 * j, 40, 0, lane);
        #pragma unroll
        for (int i = 0; i < 2; i++)
            #pragma unroll
            for (int j = 0; j < 4; j++)
                c[i][j] = __builtin_amdgcn_wmma_f32_16x16x32_bf16(
                    false, af[i], false, bfr[j], (short)0, c[i][j], false, false);
        if (it + 1 < NIT) {
            wait_async0();
            __syncthreads();
        }
    }

    // epilogue: C layout: VGPR e -> row 8*(lane/16)+e, col lane%16
    #pragma unroll
    for (int i = 0; i < 2; i++) {
        #pragma unroll
        for (int j = 0; j < 4; j++) {
            int rbase = mb + 32 * wm + 16 * i + 8 * (lane >> 4);
            int col   = nb + 64 * wn + 16 * j + (lane & 15);
            float badd = 0.f;
            if constexpr (HAS_BIAS) badd = bias[col];
            #pragma unroll
            for (int e = 0; e < 8; e++) {
                int row = rbase + e;
                float v = c[i][j][e] + badd;
                if constexpr (HAS_RES) v += res[(size_t)row * N + col];
                outF[(size_t)row * N + col] = v;
            }
        }
    }
}

// ---------------------------------------------------------------------------
// Fused SwiGLU GEMM: a-tile and g-tile of w_h4h computed together,
// writes silu(a)*g bf16. Same async/double-buffer pipeline.
// ---------------------------------------------------------------------------
__global__ __launch_bounds__(256)
void gemm_swiglu_kernel(const bf16* __restrict__ A, const float* __restrict__ W,
                        bf16* __restrict__ out, int M, int Nh, int K) {
    __shared__ bf16 As[2][128 * 40];
    __shared__ bf16 B1[2][128 * 40];
    __shared__ bf16 B2[2][128 * 40];
    const int tid = threadIdx.x, lane = tid & 31, w = tid >> 5;
    const int wm = w & 3, wn = w >> 2;
    const int mb = blockIdx.x * 128, nb = blockIdx.y * 128;
    const int NIT = K >> 5;

    const int ar0 = tid >> 2,          akc0 = (tid & 3) << 3;
    const int ar1 = (tid + 256) >> 2,  akc1 = ((tid + 256) & 3) << 3;
    unsigned int ldsA0[2], ldsA1[2];
    for (int b = 0; b < 2; b++) {
        ldsA0[b] = (unsigned int)(uintptr_t)&As[b][ar0 * 40 + akc0];
        ldsA1[b] = (unsigned int)(uintptr_t)&As[b][ar1 * 40 + akc1];
    }
    const unsigned long long abase = (unsigned long long)(uintptr_t)A;
    const unsigned int ag0 = (unsigned int)(((size_t)(mb + ar0) * K + akc0) * 2);
    const unsigned int ag1 = (unsigned int)(((size_t)(mb + ar1) * K + akc1) * 2);

    auto issueA = [&](int it, int buf) {
        const unsigned int kb = (unsigned int)it << 6;
        async_ld16(ldsA0[buf], ag0 + kb, abase);
        async_ld16(ldsA1[buf], ag1 + kb, abase);
    };

    float4 breg1[4], breg2[4];
    auto loadB = [&](int it) {
        const int k0 = it << 5;
        #pragma unroll
        for (int i = 0; i < 4; i++) {
            int ch = tid + 256 * i;
            int r = ch >> 3, kc = (ch & 7) << 2;
            breg1[i] = *(const float4*)&W[(size_t)(nb + r) * K + k0 + kc];
            breg2[i] = *(const float4*)&W[(size_t)(Nh + nb + r) * K + k0 + kc];
        }
    };
    auto storeB = [&](int buf) {
        #pragma unroll
        for (int i = 0; i < 4; i++) {
            int ch = tid + 256 * i;
            int r = ch >> 3, kc = (ch & 7) << 2;
            bf16* d1 = &B1[buf][r * 40 + kc];
            d1[0] = (bf16)breg1[i].x; d1[1] = (bf16)breg1[i].y;
            d1[2] = (bf16)breg1[i].z; d1[3] = (bf16)breg1[i].w;
            bf16* d2 = &B2[buf][r * 40 + kc];
            d2[0] = (bf16)breg2[i].x; d2[1] = (bf16)breg2[i].y;
            d2[2] = (bf16)breg2[i].z; d2[3] = (bf16)breg2[i].w;
        }
    };

    v8f c1[2][4], c2[2][4];
    #pragma unroll
    for (int i = 0; i < 2; i++)
        for (int j = 0; j < 4; j++)
            for (int e = 0; e < 8; e++) { c1[i][j][e] = 0.f; c2[i][j][e] = 0.f; }

    issueA(0, 0);
    loadB(0);
    storeB(0);
    wait_async0();
    __syncthreads();

    for (int it = 0; it < NIT; ++it) {
        const int cur = it & 1;
        if (it + 1 < NIT) {
            issueA(it + 1, cur ^ 1);
            loadB(it + 1);
            storeB(cur ^ 1);
        }
        v16bf af[2];
        #pragma unroll
        for (int i = 0; i < 2; i++) af[i] = ld_frag(As[cur], 32 * wm + 16 * i, 40, 0, lane);
        #pragma unroll
        for (int j = 0; j < 4; j++) {
            v16bf b1 = ld_frag(B1[cur], 64 * wn + 16 * j, 40, 0, lane);
            v16bf b2 = ld_frag(B2[cur], 64 * wn + 16 * j, 40, 0, lane);
            #pragma unroll
            for (int i = 0; i < 2; i++) {
                c1[i][j] = __builtin_amdgcn_wmma_f32_16x16x32_bf16(
                    false, af[i], false, b1, (short)0, c1[i][j], false, false);
                c2[i][j] = __builtin_amdgcn_wmma_f32_16x16x32_bf16(
                    false, af[i], false, b2, (short)0, c2[i][j], false, false);
            }
        }
        if (it + 1 < NIT) {
            wait_async0();
            __syncthreads();
        }
    }

    #pragma unroll
    for (int i = 0; i < 2; i++) {
        #pragma unroll
        for (int j = 0; j < 4; j++) {
            int rbase = mb + 32 * wm + 16 * i + 8 * (lane >> 4);
            int col   = nb + 64 * wn + 16 * j + (lane & 15);
            #pragma unroll
            for (int e = 0; e < 8; e++) {
                int row = rbase + e;
                float a = c1[i][j][e];
                float g = c2[i][j][e];
                // silu(a) = a / (1 + e^-a), via v_exp_f32 + v_rcp_f32
                float s = a * fast_rcp(1.f + fast_exp2(-a * LOG2E));
                out[(size_t)row * Nh + col] = (bf16)(s * g);
            }
        }
    }
}

// ---------------------------------------------------------------------------
// RoPE (bias already applied in GEMM); split mixed -> per-head-major bf16 Q/K/V
// grid (2048, 34): y<32 => Q head, y>=32 => KV head (rope K, plain V). block 128.
// ---------------------------------------------------------------------------
__global__ __launch_bounds__(128)
void rope_kernel(const float* __restrict__ mixed, const float* __restrict__ cache,
                 bf16* __restrict__ qb, bf16* __restrict__ kb, bf16* __restrict__ vb) {
    const int s = blockIdx.x, h = blockIdx.y, tid = threadIdx.x;
    const float* cc = cache + (size_t)s * 64;
    if (h < 32) {
        const float* in = mixed + (size_t)s * 4608 + h * 128;
        bf16* out = qb + ((size_t)h * 2048 + s) * 128;
        if (tid < 32) {
            float c0 = cc[2 * tid], c1 = cc[2 * tid + 1];
            float x0 = in[2 * tid], x1 = in[2 * tid + 1];
            out[2 * tid]     = (bf16)(x0 * c0 - x1 * c1);
            out[2 * tid + 1] = (bf16)(x1 * c0 + x0 * c1);
        } else if (tid >= 64) {
            out[tid] = (bf16)in[tid];
        }
    } else {
        const int kv = h - 32;
        const float* in = mixed + (size_t)s * 4608 + 4096 + kv * 128;
        bf16* out = kb + ((size_t)kv * 2048 + s) * 128;
        if (tid < 32) {
            float c0 = cc[2 * tid], c1 = cc[2 * tid + 1];
            float x0 = in[2 * tid], x1 = in[2 * tid + 1];
            out[2 * tid]     = (bf16)(x0 * c0 - x1 * c1);
            out[2 * tid + 1] = (bf16)(x1 * c0 + x0 * c1);
        } else if (tid >= 64) {
            out[tid] = (bf16)in[tid];
        }
        const float* inv = mixed + (size_t)s * 4608 + 4352 + kv * 128;
        vb[((size_t)kv * 2048 + s) * 128 + tid] = (bf16)inv[tid];
    }
}

// ---------------------------------------------------------------------------
// Flash attention with causal mask + GQA (16 Q heads per KV head).
// One (head, 128-query-tile) per WG. 8 waves, wave owns 16 query rows.
// 32 KB LDS: Q staged then region reused as K/P (16KB) + V^T (16KB).
// ---------------------------------------------------------------------------
__global__ __launch_bounds__(256)
void flash_attn_kernel(const bf16* __restrict__ q, const bf16* __restrict__ k,
                       const bf16* __restrict__ v, bf16* __restrict__ ctx) {
    __shared__ bf16 smem[128 * 128];          // 32 KB
    bf16* Qs = smem;                          // Q stage (dead after frag preload)
    bf16* KP = smem;                          // K tile [64][128] then P [128][64]
    bf16* VT = smem + 64 * 128;               // V^T [128 dims][64 keys]

    const int tid = threadIdx.x, lane = tid & 31, w = tid >> 5;
    const int qt = blockIdx.x, head = blockIdx.y;
    const int qbase = qt * 128;
    const int kvh = head >> 4;
    const bf16* qh = q + (size_t)head * 2048 * 128;
    const bf16* kh = k + (size_t)kvh * 2048 * 128;
    const bf16* vh = v + (size_t)kvh * 2048 * 128;

    // stage Q tile 128x128
    for (int i = 0; i < 8; i++) {
        int ch = tid + 256 * i;
        int r = ch >> 4, kc = (ch & 15) << 3;
        *(v8bf*)&Qs[r * 128 + kc] = *(const v8bf*)&qh[(size_t)(qbase + r) * 128 + kc];
    }
    __syncthreads();
    v16bf qf[4];
    for (int ks = 0; ks < 4; ks++) qf[ks] = ld_frag(Qs, 16 * w, 128, 32 * ks, lane);

    v8f acc[8];
    for (int t = 0; t < 8; t++)
        for (int e = 0; e < 8; e++) acc[t][e] = 0.f;
    float mrow[8], lrow[8];
    for (int j = 0; j < 8; j++) { mrow[j] = -1e30f; lrow[j] = 0.f; }

    const float scale = 0.08838834764831845f;   // 1/sqrt(128)
    const int rbase = qbase + 16 * w + 8 * (lane >> 4);
    const int nkt = qt * 2 + 2;                 // key tiles of 64 up to diag

    for (int kt = 0; kt < nkt; kt++) {
        const int kb = kt * 64;
        __syncthreads();   // smem free (Q frags preloaded / prev P,V consumed)
        for (int i = 0; i < 4; i++) {           // stage K [64][128]
            int ch = tid + 256 * i;
            int r = ch >> 4, kc = (ch & 15) << 3;
            *(v8bf*)&KP[r * 128 + kc] = *(const v8bf*)&kh[(size_t)(kb + r) * 128 + kc];
        }
        for (int i = 0; i < 4; i++) {           // stage V^T [128][64]
            int ch = tid + 256 * i;
            int r = ch >> 4, kc = (ch & 15) << 3;
            v8bf vv = *(const v8bf*)&vh[(size_t)(kb + r) * 128 + kc];
            for (int e = 0; e < 8; e++) VT[(kc + e) * 64 + r] = vv[e];
        }
        __syncthreads();

        // S = Q K^T  (16 rows x 64 keys per wave)
        v8f s[4];
        for (int nt = 0; nt < 4; nt++) {
            v8f sc;
            for (int e = 0; e < 8; e++) sc[e] = 0.f;
            for (int ks = 0; ks < 4; ks++) {
                v16bf b = ld_frag(KP, 16 * nt, 128, 32 * ks, lane);
                sc = __builtin_amdgcn_wmma_f32_16x16x32_bf16(
                    false, qf[ks], false, b, (short)0, sc, false, false);
            }
            s[nt] = sc;
        }
        __syncthreads();   // K reads done; KP reusable for P

        // scale + causal mask
        for (int nt = 0; nt < 4; nt++) {
            int col = kb + 16 * nt + (lane & 15);
            for (int j = 0; j < 8; j++) {
                float sv = s[nt][j] * scale;
                if (col > rbase + j) sv = -1e30f;
                s[nt][j] = sv;
            }
        }
        // online softmax row stats (16-lane-half shuffle reductions)
        float corr[8];
        for (int j = 0; j < 8; j++) {
            float mx = -1e30f;
            for (int nt = 0; nt < 4; nt++) mx = fmaxf(mx, s[nt][j]);
            for (int off = 8; off > 0; off >>= 1) mx = fmaxf(mx, __shfl_xor(mx, off, 32));
            float mn = fmaxf(mrow[j], mx);
            corr[j] = fast_exp2((mrow[j] - mn) * LOG2E);
            mrow[j] = mn;
        }
        float rs[8];
        for (int j = 0; j < 8; j++) rs[j] = 0.f;
        for (int nt = 0; nt < 4; nt++)
            for (int j = 0; j < 8; j++) {
                float pv = (s[nt][j] <= -1e29f) ? 0.f
                                                : fast_exp2((s[nt][j] - mrow[j]) * LOG2E);
                s[nt][j] = pv;
                rs[j] += pv;
            }
        for (int j = 0; j < 8; j++) {
            float t = rs[j];
            for (int off = 8; off > 0; off >>= 1) t += __shfl_xor(t, off, 32);
            lrow[j] = lrow[j] * corr[j] + t;
            for (int tt = 0; tt < 8; tt++) acc[tt][j] *= corr[j];
        }
        // write P (bf16) into KP as [128 rows][64 keys]
        for (int nt = 0; nt < 4; nt++) {
            int colL = 16 * nt + (lane & 15);
            for (int j = 0; j < 8; j++) {
                int rowL = 16 * w + 8 * (lane >> 4) + j;
                KP[rowL * 64 + colL] = (bf16)s[nt][j];
            }
        }
        __syncthreads();
        // O += P V
        for (int ks = 0; ks < 2; ks++) {
            v16bf a = ld_frag(KP, 16 * w, 64, 32 * ks, lane);
            for (int nt = 0; nt < 8; nt++) {
                v16bf b = ld_frag(VT, 16 * nt, 64, 32 * ks, lane);
                acc[nt] = __builtin_amdgcn_wmma_f32_16x16x32_bf16(
                    false, a, false, b, (short)0, acc[nt], false, false);
            }
        }
    }

    // epilogue: one v_rcp per row, then multiply; ctx[s][head*128 + d] bf16
    float linv[8];
    for (int j = 0; j < 8; j++) linv[j] = fast_rcp(lrow[j]);
    for (int t = 0; t < 8; t++) {
        int dim = 16 * t + (lane & 15);
        for (int j = 0; j < 8; j++) {
            int row = qbase + 16 * w + 8 * (lane >> 4) + j;
            float val = acc[t][j] * linv[j];
            ctx[(size_t)row * 4096 + head * 128 + dim] = (bf16)val;
        }
    }
}

// ---------------------------------------------------------------------------
extern "C" void kernel_launch(void* const* d_in, const int* in_sizes, int n_in,
                              void* d_out, int out_size, void* d_ws, size_t ws_size,
                              hipStream_t stream) {
    const float* hidden   = (const float*)d_in[0];   // (2048,1,4096)
    const float* ropec    = (const float*)d_in[1];   // (2048,32,2)
    const float* w_ln1    = (const float*)d_in[2];   // (4096,)
    const float* w_qkv    = (const float*)d_in[3];   // (4608,4096)
    const float* b_qkv    = (const float*)d_in[4];   // (4608,)
    const float* w_dense  = (const float*)d_in[5];   // (4096,4096)
    const float* w_ln2    = (const float*)d_in[6];   // (4096,)
    const float* w_h4h    = (const float*)d_in[7];   // (27392,4096)
    const float* w_4hh    = (const float*)d_in[8];   // (4096,13696)
    float* out = (float*)d_out;

    const int SQ = 2048, H = 4096, NQKV = 4608, FF = 13696;

    char* ws = (char*)d_ws;
    size_t off = 0;
    auto take = [&](size_t bytes) -> char* {
        char* p = ws + off;
        off = (off + bytes + 255) & ~(size_t)255;
        return p;
    };
    bf16*  ln1   = (bf16*) take((size_t)SQ * H * 2);
    float* mixed = (float*)take((size_t)SQ * NQKV * 4);
    bf16*  qb    = (bf16*) take((size_t)32 * SQ * 128 * 2);
    bf16*  kb    = (bf16*) take((size_t)2 * SQ * 128 * 2);
    bf16*  vb    = (bf16*) take((size_t)2 * SQ * 128 * 2);
    bf16*  ctx   = (bf16*) take((size_t)SQ * H * 2);
    float* ln_in = (float*)take((size_t)SQ * H * 4);
    bf16*  ln2   = (bf16*) take((size_t)SQ * H * 2);
    bf16*  act   = (bf16*) take((size_t)SQ * FF * 2);
    (void)ws_size; (void)n_in; (void)in_sizes; (void)out_size;

    // 1. ln1 = rmsnorm(hidden) -> bf16
    rmsnorm_kernel<<<SQ, 256, 0, stream>>>(hidden, w_ln1, ln1, H);
    // 2. mixed = ln1 @ w_qkv^T + b_qkv  (fp32 out)
    gemm_bf16_kernel<true, false><<<dim3(SQ / 128, NQKV / 128), 256, 0, stream>>>(
        ln1, w_qkv, b_qkv, nullptr, mixed, SQ, NQKV, H);
    // 3. RoPE + split -> per-head bf16 q/k/v
    rope_kernel<<<dim3(SQ, 34), 128, 0, stream>>>(mixed, ropec, qb, kb, vb);
    // 4. causal GQA flash attention -> ctx bf16 [sq][4096]
    flash_attn_kernel<<<dim3(SQ / 128, 32), 256, 0, stream>>>(qb, kb, vb, ctx);
    // 5. ln_in = ctx @ w_dense^T + hidden  (fp32)
    gemm_bf16_kernel<false, true><<<dim3(SQ / 128, H / 128), 256, 0, stream>>>(
        ctx, w_dense, nullptr, hidden, ln_in, SQ, H, H);
    // 6. ln2 = rmsnorm(ln_in) -> bf16
    rmsnorm_kernel<<<SQ, 256, 0, stream>>>(ln_in, w_ln2, ln2, H);
    // 7. act = silu(ln2 @ Wa^T) * (ln2 @ Wg^T) -> bf16
    gemm_swiglu_kernel<<<dim3(SQ / 128, FF / 128), 256, 0, stream>>>(
        ln2, w_h4h, act, SQ, FF, H);
    // 8. out = act @ w_4hh^T + ln_in  (fp32)
    gemm_bf16_kernel<false, true><<<dim3(SQ / 128, H / 128), 256, 0, stream>>>(
        act, w_4hh, nullptr, ln_in, out, SQ, H, FF);
}
